// TrackMamba_19722489823902
// MI455X (gfx1250) — matrix operand order repaired and checked by reading, not compiled, not run on hardware
//
#include <hip/hip_runtime.h>
#include <hip/hip_bf16.h>

typedef __attribute__((ext_vector_type(16))) _Float16 v16h;
typedef __attribute__((ext_vector_type(8)))  _Float16 v8h;
typedef __attribute__((ext_vector_type(8)))  float    v8f;

// ---------------- problem constants ----------------
constexpr int GEOM = 8, QUAL = 7, TIMEF = 4, CTX = 4, HID = 128;
constexpr int Bb = 8, Tt = 128, Ll = 64, Dd = 128;
constexpr int NS = Bb * Tt;          // 1024 sequences
constexpr float EPS = 1e-5f;

// packed scan-weight region: per block, layers [mw,mc,ow,oc,rw,rc,out]
// K dims {256,384,256,384,256,384,384}; fragment = 512 halves (32 lanes x 16)
constexpr int BLK_PACK_HALVES = 294912;              // (3*256+4*384)*128
constexpr int PACK_REG1 = 2 * BLK_PACK_HALVES;       // both blocks
constexpr int PACK_TOTAL = PACK_REG1 + 2 * 16384;    // + W1d, W1f (128x128 each)
// per-layer half offsets within a block pack
__device__ __constant__ int c_layer_off[8] = {0, 32768, 81920, 131072, 163840, 212992, 262144, 294912};
// v16h-unit offsets (half offset / 16)
constexpr int FO_MW = 0, FO_MC = 2048, FO_OW = 5120, FO_OC = 8192,
              FO_RW = 10240, FO_RC = 13312, FO_OUT = 16384;

// ---------------- helpers ----------------
__device__ __forceinline__ float sigm(float x) { return 1.0f / (1.0f + expf(-x)); }

// One 16x16x32 WMMA K-step. A comes from an LDS tile (row-major, `stride`
// halves per row), rows are the 16 M entries; B comes from pre-packed global
// fragments (per-lane contiguous 16 halves -> two global_load_b128).
__device__ __forceinline__ v8f wstep(v8f acc, const _Float16* tileRow0, int stride,
                                     int kbase, const v16h* wbase, int frag, int lane) {
  const _Float16* ap = tileRow0 + (size_t)(lane & 15) * stride + kbase + ((lane >> 4) << 4);
  v8h lo = *(const v8h*)ap;          // K = kbase + hi*16 .. +7   (ds_load_b128)
  v8h hi = *(const v8h*)(ap + 8);    // K = kbase + hi*16 + 8 .. +15
  v16h a;
#pragma unroll
  for (int i = 0; i < 8; ++i) { a[i] = lo[i]; a[i + 8] = hi[i]; }
  v16h bf = wbase[frag * 32 + lane];
  return __builtin_amdgcn_wmma_f32_16x16x32_f16(false, a, false, bf, (short)0, acc, false, false);
}

// ---------------- weight packing ----------------
struct PackSrc { const float* w[2][7]; const float* pair1; };

__global__ void k_pack(PackSrc ps, _Float16* __restrict__ dst) {
  int idx = blockIdx.x * 256 + threadIdx.x;
  if (idx >= PACK_TOTAL) return;
  if (idx < PACK_REG1) {
    int blk = idx / BLK_PACK_HALVES;
    int r = idx - blk * BLK_PACK_HALVES;
    int layer = 0;
    while (r >= c_layer_off[layer + 1]) ++layer;
    int h = r - c_layer_off[layer];
    int frag = h >> 9, lane = (h >> 4) & 31, i = h & 15;
    int nt = frag & 7, kt = frag >> 3;
    int n = nt * 16 + (lane & 15);
    int k = kt * 32 + ((lane >> 4) * 16) + i;   // B-operand: k = (lane>=16)*16 + i within chunk
    dst[idx] = (_Float16)ps.w[blk][layer][k * HID + n];
  } else {
    int r = idx - PACK_REG1;
    int sel = r / 16384, h = r & 16383;
    int frag = h >> 9, lane = (h >> 4) & 31, i = h & 15;
    int nt = frag & 7, kt = frag >> 3;
    int n = nt * 16 + (lane & 15);
    int k = kt * 32 + ((lane >> 4) * 16) + i;
    int kofs = (sel == 0) ? 384 : 640;          // W1d rows / W1f rows of pair1
    dst[idx] = (_Float16)ps.pair1[(kofs + k) * HID + n];
  }
}

// ---------------- block prologue: LN + input linear ----------------
__global__ __launch_bounds__(256)
void k_inlin(int blk, int inDim,
             const float* __restrict__ geom, const float* __restrict__ qual,
             const float* __restrict__ timef, const _Float16* __restrict__ yprev,
             const float* __restrict__ lng, const float* __restrict__ lnb,
             const float* __restrict__ Wi, const float* __restrict__ bi,
             _Float16* __restrict__ dst) {
  size_t nl = (size_t)blockIdx.x * 256 + threadIdx.x;
  if (nl >= (size_t)NS * Ll) return;
  float x[128];
  if (blk == 0) {
    for (int k = 0; k < GEOM; ++k)  x[k] = geom[nl * GEOM + k];
    for (int k = 0; k < QUAL; ++k)  x[GEOM + k] = qual[nl * QUAL + k];
    for (int k = 0; k < TIMEF; ++k) x[GEOM + QUAL + k] = timef[nl * TIMEF + k];
  } else {
    for (int k = 0; k < HID; ++k) x[k] = (float)yprev[nl * HID + k];
  }
  float mu = 0.f;
  for (int k = 0; k < inDim; ++k) mu += x[k];
  mu /= (float)inDim;
  float var = 0.f;
  for (int k = 0; k < inDim; ++k) { float d = x[k] - mu; var += d * d; }
  var /= (float)inDim;
  float inv = rsqrtf(var + EPS);
  for (int k = 0; k < inDim; ++k) x[k] = (x[k] - mu) * inv * lng[k] + lnb[k];
  for (int c = 0; c < HID; ++c) {
    float a = bi[c];
    for (int k = 0; k < inDim; ++k) a += x[k] * Wi[k * HID + c];
    dst[nl * HID + c] = (_Float16)a;
  }
}

// depthwise conv1d k=3, pad=1 along L
__global__ __launch_bounds__(256)
void k_conv(const _Float16* __restrict__ src, _Float16* __restrict__ dst,
            const float* __restrict__ Wc, const float* __restrict__ bc) {
  size_t i = (size_t)blockIdx.x * 256 + threadIdx.x;
  if (i >= (size_t)NS * Ll * HID) return;
  int c = (int)(i % HID);
  size_t nl = i / HID;
  int l = (int)(nl % Ll);
  size_t n = nl / Ll;
  float acc = bc[c];
#pragma unroll
  for (int j = 0; j < 3; ++j) {
    int ls = l - 1 + j;
    if (ls >= 0 && ls < Ll) acc += Wc[c * 3 + j] * (float)src[(n * Ll + ls) * HID + c];
  }
  dst[i] = (_Float16)acc;
}

__global__ __launch_bounds__(256)
void k_qlin(const float* __restrict__ qual, const float* __restrict__ Wq,
            const float* __restrict__ bq, _Float16* __restrict__ dst) {
  size_t nl = (size_t)blockIdx.x * 256 + threadIdx.x;
  if (nl >= (size_t)NS * Ll) return;
  float q[QUAL];
  for (int k = 0; k < QUAL; ++k) q[k] = qual[nl * QUAL + k];
  for (int c = 0; c < HID; ++c) {
    float a = bq[c];
    for (int k = 0; k < QUAL; ++k) a += q[k] * Wq[k * HID + c];
    dst[nl * HID + c] = (_Float16)a;
  }
}

// ---------------- WMMA gated SSM scan ----------------
struct ScanBias { const float* b[7]; };   // mw,mc,ow,oc,rw,rc,out biases

__global__ __launch_bounds__(256)
void k_scan(const _Float16* __restrict__ hbuf, const _Float16* __restrict__ qbuf,
            _Float16* __restrict__ ybuf, const _Float16* __restrict__ wpk,
            ScanBias bias, const float* __restrict__ mask,
            float* __restrict__ mOut, float* __restrict__ oOut, float* __restrict__ rOut) {
  __shared__ _Float16 s_xq[16][256];                 // [x_t | q_t] f16
  __shared__ _Float16 s_m16[16][128], s_o16[16][128], s_r16[16][128];
  __shared__ float    s_mS[16][128], s_oS[16][128], s_rS[16][128];
  __shared__ float    s_mask[16][Ll];                // mask tile staged once
  const int tid = threadIdx.x, lane = tid & 31, w = tid >> 5;   // wave w owns column tile w
  const int n0 = blockIdx.x * 16;
  for (int e = tid; e < 16 * 128; e += 256) {
    int r = e >> 7, c = e & 127;
    s_mS[r][c] = 0.f; s_oS[r][c] = 0.f; s_rS[r][c] = 0.f;
    s_m16[r][c] = (_Float16)0.f; s_o16[r][c] = (_Float16)0.f; s_r16[r][c] = (_Float16)0.f;
  }
  for (int e = tid; e < 16 * Ll; e += 256) {
    int r = e / Ll, t = e % Ll;
    s_mask[r][t] = mask[(size_t)(n0 + r) * Ll + t];
  }
  __syncthreads();
  const v16h* wp = (const v16h*)wpk;
  const _Float16* xT = &s_xq[0][0];
  const _Float16* mT = &s_m16[0][0];
  const _Float16* oT = &s_o16[0][0];
  const _Float16* rT = &s_r16[0][0];
  const int col = w * 16 + (lane & 15);
  const int rbase = (lane < 16) ? 0 : 8;
  const v8f vz = {0.f, 0.f, 0.f, 0.f, 0.f, 0.f, 0.f, 0.f};

#pragma unroll 1
  for (int t = 0; t < Ll; ++t) {
    for (int e = tid; e < 16 * 256; e += 256) {
      int r = e >> 8, c = e & 255;
      s_xq[r][c] = (c < 128) ? hbuf[((size_t)(n0 + r) * Ll + t) * HID + c]
                             : qbuf[((size_t)(n0 + r) * Ll + t) * HID + (c - 128)];
    }
    __syncthreads();
    // Block LICM: without this, the compiler hoists all loop-invariant weight
    // fragments out of the t-loop, blowing past 256 VGPRs and spilling every
    // B operand to scratch. Keep them as per-iteration L2 loads instead.
    asm volatile("" ::: "memory");
    // ---- six gate GEMMs (reads previous-step states) ----
    v8f amw = vz, amc = vz, aow = vz, aoc = vz, arw = vz, arc = vz;
#pragma unroll
    for (int kt = 0; kt < 8; ++kt) amw = wstep(amw, xT, 256, kt * 32, wp + FO_MW, kt * 8 + w, lane);
#pragma unroll
    for (int kt = 0; kt < 8; ++kt) amc = wstep(amc, xT, 256, kt * 32, wp + FO_MC, kt * 8 + w, lane);
#pragma unroll
    for (int kt = 0; kt < 4; ++kt) amc = wstep(amc, oT, 128, kt * 32, wp + FO_MC, (8 + kt) * 8 + w, lane);
#pragma unroll
    for (int kt = 0; kt < 4; ++kt) aow = wstep(aow, xT, 256, 128 + kt * 32, wp + FO_OW, kt * 8 + w, lane);
#pragma unroll
    for (int kt = 0; kt < 4; ++kt) aow = wstep(aow, mT, 128, kt * 32, wp + FO_OW, (4 + kt) * 8 + w, lane);
#pragma unroll
    for (int kt = 0; kt < 4; ++kt) aoc = wstep(aoc, xT, 256, 128 + kt * 32, wp + FO_OC, kt * 8 + w, lane);
#pragma unroll
    for (int kt = 0; kt < 4; ++kt) aoc = wstep(aoc, mT, 128, kt * 32, wp + FO_OC, (4 + kt) * 8 + w, lane);
#pragma unroll
    for (int kt = 0; kt < 4; ++kt) aoc = wstep(aoc, rT, 128, kt * 32, wp + FO_OC, (8 + kt) * 8 + w, lane);
#pragma unroll
    for (int kt = 0; kt < 4; ++kt) arw = wstep(arw, xT, 256, 128 + kt * 32, wp + FO_RW, kt * 8 + w, lane);
#pragma unroll
    for (int kt = 0; kt < 4; ++kt) arw = wstep(arw, oT, 128, kt * 32, wp + FO_RW, (4 + kt) * 8 + w, lane);
#pragma unroll
    for (int kt = 0; kt < 4; ++kt) arc = wstep(arc, xT, 256, kt * 32, wp + FO_RC, kt * 8 + w, lane);
#pragma unroll
    for (int kt = 0; kt < 4; ++kt) arc = wstep(arc, mT, 128, kt * 32, wp + FO_RC, (4 + kt) * 8 + w, lane);
#pragma unroll
    for (int kt = 0; kt < 4; ++kt) arc = wstep(arc, oT, 128, kt * 32, wp + FO_RC, (8 + kt) * 8 + w, lane);
    const float bmw = bias.b[0][col], bmc = bias.b[1][col], bow = bias.b[2][col],
                boc = bias.b[3][col], brw = bias.b[4][col], brc = bias.b[5][col];
#pragma unroll
    for (int v = 0; v < 8; ++v) {
      amw[v] = sigm(amw[v] + bmw);  amc[v] = tanhf(amc[v] + bmc);
      aow[v] = sigm(aow[v] + bow);  aoc[v] = tanhf(aoc[v] + boc);
      arw[v] = sigm(arw[v] + brw);  arc[v] = tanhf(arc[v] + brc);
    }
    __syncthreads();   // all waves done reading old states
    // ---- masked state update (each wave updates its own columns) ----
#pragma unroll
    for (int v = 0; v < 8; ++v) {
      int row = rbase + v;
      float mk = s_mask[row][t];
      float m = s_mS[row][col], o = s_oS[row][col], r = s_rS[row][col];
      float nm = (1.f - amw[v]) * m + amw[v] * amc[v];
      float no = (1.f - aow[v]) * o + aow[v] * aoc[v];
      float nr = (1.f - arw[v]) * r + arw[v] * arc[v];
      if (mk > 0.f) { m = nm; o = no; r = nr; }
      s_mS[row][col] = m; s_oS[row][col] = o; s_rS[row][col] = r;
      s_m16[row][col] = (_Float16)m; s_o16[row][col] = (_Float16)o; s_r16[row][col] = (_Float16)r;
    }
    __syncthreads();
    // ---- output projection on new states, residual +h ----
    v8f ao = vz;
#pragma unroll
    for (int kt = 0; kt < 4; ++kt) ao = wstep(ao, mT, 128, kt * 32, wp + FO_OUT, kt * 8 + w, lane);
#pragma unroll
    for (int kt = 0; kt < 4; ++kt) ao = wstep(ao, oT, 128, kt * 32, wp + FO_OUT, (4 + kt) * 8 + w, lane);
#pragma unroll
    for (int kt = 0; kt < 4; ++kt) ao = wstep(ao, rT, 128, kt * 32, wp + FO_OUT, (8 + kt) * 8 + w, lane);
    const float bo = bias.b[6][col];
#pragma unroll
    for (int v = 0; v < 8; ++v) {
      int row = rbase + v;
      float y = ao[v] + bo + (float)s_xq[row][col];
      ybuf[((size_t)(n0 + row) * Ll + t) * HID + col] = (_Float16)y;
    }
    if (t + 1 < Ll)
      __builtin_prefetch(&hbuf[((size_t)(n0 + (tid & 15)) * Ll + (t + 1)) * HID + ((tid >> 4) << 3)], 0, 1);
    __syncthreads();
  }
  for (int e = tid; e < 16 * 128; e += 256) {
    int r = e >> 7, c = e & 127;
    mOut[(size_t)(n0 + r) * HID + c] = s_mS[r][c];
    oOut[(size_t)(n0 + r) * HID + c] = s_oS[r][c];
    rOut[(size_t)(n0 + r) * HID + c] = s_rS[r][c];
  }
}

// ---------------- final LN + masked mean pool ----------------
__global__ __launch_bounds__(64)
void k_pool(const _Float16* __restrict__ y, const float* __restrict__ mask,
            const float* __restrict__ g, const float* __restrict__ be,
            float* __restrict__ pooled) {
  int n = blockIdx.x, l = threadIdx.x;
  __shared__ float sp[HID];
  __shared__ float scnt;
  for (int c = l; c < HID; c += 64) sp[c] = 0.f;
  if (l == 0) scnt = 0.f;
  __syncthreads();
  const _Float16* row = y + ((size_t)n * Ll + l) * HID;
  float mu = 0.f;
  for (int c = 0; c < HID; ++c) mu += (float)row[c];
  mu /= (float)HID;
  float var = 0.f;
  for (int c = 0; c < HID; ++c) { float d = (float)row[c] - mu; var += d * d; }
  var /= (float)HID;
  float inv = rsqrtf(var + EPS);
  float mk = mask[(size_t)n * Ll + l];
  for (int c = 0; c < HID; ++c) {
    float val = ((float)row[c] - mu) * inv * g[c] + be[c];
    atomicAdd(&sp[c], val * mk);
  }
  atomicAdd(&scnt, mk);
  __syncthreads();
  float denom = fmaxf(scnt, 1.f);
  for (int c = l; c < HID; c += 64) pooled[(size_t)n * HID + c] = sp[c] / denom;
}

// ---------------- per-track heads + pair precompute c_t ----------------
struct TrackArgs {
  const float *pooled, *mS, *oS, *rS, *ctx;
  const float *ctx1W, *ctx1b, *ctx2W, *ctx2b;
  const float *f1W, *f1b, *f2W, *f2b;
  const float *ns1W, *ns1b, *ns2W, *ns2b;
  const float *rec1W, *rec1b, *rec2W, *rec2b;
  const float *sur1W, *sur1b, *sur2W, *sur2b;
  const float *gpW, *gpb, *p1W, *p1b;
  float *outRec, *outSur, *outNs, *c_t;
};

__global__ __launch_bounds__(128) void k_track(TrackArgs A) {
  int n = blockIdx.x, c = threadIdx.x;
  __shared__ float s_in[4 * HID];
  __shared__ float s_h[HID], s_tr[HID], s_r[HID], s_pne[HID];
  __shared__ float s_ns[GEOM];
  s_in[c] = A.pooled[(size_t)n * HID + c];
  s_in[HID + c] = A.mS[(size_t)n * HID + c];
  s_in[2 * HID + c] = A.oS[(size_t)n * HID + c];
  s_r[c] = A.rS[(size_t)n * HID + c];
  float acc = A.ctx1b[c];
  for (int k = 0; k < CTX; ++k) acc += A.ctx[n * CTX + k] * A.ctx1W[k * HID + c];
  s_h[c] = fmaxf(acc, 0.f);
  __syncthreads();
  acc = A.ctx2b[c];
  for (int k = 0; k < HID; ++k) acc += s_h[k] * A.ctx2W[k * HID + c];
  __syncthreads();
  s_in[3 * HID + c] = acc;
  __syncthreads();
  acc = A.f1b[c];
  for (int k = 0; k < 4 * HID; ++k) acc += s_in[k] * A.f1W[k * HID + c];
  float fh = fmaxf(acc, 0.f);
  __syncthreads(); s_h[c] = fh; __syncthreads();
  acc = A.f2b[c];
  for (int k = 0; k < HID; ++k) acc += s_h[k] * A.f2W[k * HID + c];
  s_tr[c] = acc;
  __syncthreads();
  // next-state head
  acc = A.ns1b[c];
  for (int k = 0; k < HID; ++k) acc += s_tr[k] * A.ns1W[k * HID + c];
  float h1 = fmaxf(acc, 0.f);
  __syncthreads(); s_h[c] = h1; __syncthreads();
  if (c < GEOM) {
    acc = A.ns2b[c];
    for (int k = 0; k < HID; ++k) acc += s_h[k] * A.ns2W[k * GEOM + c];
    s_ns[c] = acc;
    A.outNs[(size_t)n * GEOM + c] = acc;
  }
  __syncthreads();
  // recovery head
  acc = A.rec1b[c];
  for (int k = 0; k < HID; ++k) acc += s_tr[k] * A.rec1W[k * HID + c];
  float r1 = fmaxf(acc, 0.f);
  __syncthreads(); s_h[c] = r1; __syncthreads();
  if (c < 4) {
    acc = A.rec2b[c];
    for (int k = 0; k < HID; ++k) acc += s_h[k] * A.rec2W[k * 4 + c];
    A.outRec[(size_t)n * 4 + c] = acc;
  }
  __syncthreads();
  // survival head
  float sv = 0.f;
  if (c < 64) {
    acc = A.sur1b[c];
    for (int k = 0; k < HID; ++k) acc += s_tr[k] * A.sur1W[k * 64 + c];
    sv = fmaxf(acc, 0.f);
  }
  __syncthreads(); if (c < 64) s_h[c] = sv; __syncthreads();
  if (c == 0) {
    acc = A.sur2b[0];
    for (int k = 0; k < 64; ++k) acc += s_h[k] * A.sur2W[k];
    A.outSur[n] = acc;
  }
  // pred_next_embed = geom_pair(next_state_pred)
  acc = A.gpb[c];
  for (int k = 0; k < GEOM; ++k) acc += s_ns[k] * A.gpW[k * HID + c];
  s_pne[c] = acc;
  __syncthreads();
  // c_t = tr*W1[0:128] + pne*W1[256:384] + r*W1[512:640] + b1
  float ct = A.p1b[c];
  for (int k = 0; k < HID; ++k) ct += s_tr[k] * A.p1W[k * HID + c];
  for (int k = 0; k < HID; ++k) ct += s_pne[k] * A.p1W[(2 * HID + k) * HID + c];
  for (int k = 0; k < HID; ++k) ct += s_r[k] * A.p1W[(4 * HID + k) * HID + c];
  A.c_t[(size_t)n * HID + c] = ct;
}

// ---------------- per-det: det_embed, qual_embed, c_d ----------------
__global__ __launch_bounds__(128)
void k_det(const float* __restrict__ dg, const float* __restrict__ dq,
           const float* __restrict__ d1W, const float* __restrict__ d1b,
           const float* __restrict__ d2W, const float* __restrict__ d2b,
           const float* __restrict__ qpW, const float* __restrict__ qpb,
           const float* __restrict__ gpW, const float* __restrict__ gpb,
           const float* __restrict__ p1W,
           float* __restrict__ c_d, float* __restrict__ qeOut) {
  int bd = blockIdx.x, c = threadIdx.x;
  __shared__ float s_in[GEOM + QUAL];
  __shared__ float s_h[HID], s_de[HID], s_ge[HID];
  if (c < GEOM) s_in[c] = dg[bd * GEOM + c];
  else if (c < GEOM + QUAL) s_in[c] = dq[bd * QUAL + (c - GEOM)];
  __syncthreads();
  float a = d1b[c];
  for (int k = 0; k < GEOM + QUAL; ++k) a += s_in[k] * d1W[k * HID + c];
  s_h[c] = fmaxf(a, 0.f);
  __syncthreads();
  a = d2b[c];
  for (int k = 0; k < HID; ++k) a += s_h[k] * d2W[k * HID + c];
  s_de[c] = a;
  float qe = qpb[c];
  for (int k = 0; k < QUAL; ++k) qe += s_in[GEOM + k] * qpW[k * HID + c];
  qeOut[(size_t)bd * HID + c] = qe;
  float ge = gpb[c];
  for (int k = 0; k < GEOM; ++k) ge += s_in[k] * gpW[k * HID + c];
  s_ge[c] = ge;
  __syncthreads();
  float cd = 0.f;
  for (int k = 0; k < HID; ++k) cd += s_de[k] * p1W[(HID + k) * HID + c];       // det_embed rows
  for (int k = 0; k < HID; ++k) cd -= s_ge[k] * p1W[(2 * HID + k) * HID + c];   // -det_geom_embed rows
  c_d[(size_t)bd * HID + c] = cd;
}

// ---------------- WMMA pair association kernel ----------------
__global__ __launch_bounds__(256)
void k_pair(const _Float16* __restrict__ pw4, const _Float16* __restrict__ pw6,
            const float* __restrict__ oS, const float* __restrict__ qe,
            const float* __restrict__ c_t, const float* __restrict__ c_d,
            const float* __restrict__ nsPred, const float* __restrict__ dg,
            const float* __restrict__ gpW, const float* __restrict__ gpb,
            const float* __restrict__ p2W, const float* __restrict__ p2b,
            float* __restrict__ assoc) {
  int t0 = blockIdx.x * 16, d0 = blockIdx.y * 4, b = blockIdx.z;
  int tid = threadIdx.x, lane = tid & 31, w = tid >> 5;
  __shared__ _Float16 s_u[64][128];   // u[p][k] = o_t[k]*qe_d[k]
  __shared__ _Float16 s_gd[64][128];  // geom_delta_embed
  __shared__ float s_o[16][128], s_qe[4][128];
  __shared__ float s_ns[16][GEOM], s_dg[4][GEOM];
  __shared__ float s_wg[GEOM][128], s_bg[128], s_w2[128];
  __shared__ float s_logit[64];
  for (int e = tid; e < 16 * 128; e += 256) {
    int r = e >> 7, c = e & 127;
    s_o[r][c] = oS[((size_t)(b * Tt + t0 + r)) * HID + c];
  }
  for (int e = tid; e < 4 * 128; e += 256) {
    int r = e >> 7, c = e & 127;
    s_qe[r][c] = qe[((size_t)(b * Dd + d0 + r)) * HID + c];
  }
  if (tid < 128) { int r = tid >> 3, k = tid & 7; s_ns[r][k] = nsPred[(size_t)(b * Tt + t0 + r) * GEOM + k]; }
  if (tid < 32)  { int r = tid >> 3, k = tid & 7; s_dg[r][k] = dg[(size_t)(b * Dd + d0 + r) * GEOM + k]; }
  for (int e = tid; e < GEOM * 128; e += 256) s_wg[e >> 7][e & 127] = gpW[e];
  if (tid < 128) { s_bg[tid] = gpb[tid]; s_w2[tid] = p2W[tid]; }
  if (tid < 64) s_logit[tid] = 0.f;
  __syncthreads();
  for (int e = tid; e < 64 * 128; e += 256) {
    int p = e >> 7, c = e & 127, ti = p >> 2, di = p & 3;
    s_u[p][c] = (_Float16)(s_o[ti][c] * s_qe[di][c]);
    float acc = s_bg[c];
    for (int k = 0; k < GEOM; ++k) acc += fabsf(s_dg[di][k] - s_ns[ti][k]) * s_wg[k][c];
    s_gd[p][c] = (_Float16)acc;
  }
  __syncthreads();
  const v16h* w4 = (const v16h*)pw4;
  const v16h* w6 = (const v16h*)pw6;
  int mt = w & 3, ng = w >> 2;
  int rbase = (lane < 16) ? 0 : 8, colL = lane & 15;
  const _Float16* uT = &s_u[mt * 16][0];
  const _Float16* gT = &s_gd[mt * 16][0];
  const v8f vz = {0.f, 0.f, 0.f, 0.f, 0.f, 0.f, 0.f, 0.f};
  float lp[8];
#pragma unroll
  for (int v = 0; v < 8; ++v) lp[v] = 0.f;
#pragma unroll 1
  for (int nt2 = 0; nt2 < 4; ++nt2) {
    asm volatile("" ::: "memory");   // keep weight frags as per-iteration loads
    int ntile = ng * 4 + nt2;
    v8f acc = vz;
#pragma unroll
    for (int kt = 0; kt < 4; ++kt) acc = wstep(acc, uT, 128, kt * 32, w4, kt * 8 + ntile, lane);
#pragma unroll
    for (int kt = 0; kt < 4; ++kt) acc = wstep(acc, gT, 128, kt * 32, w6, kt * 8 + ntile, lane);
    int col = ntile * 16 + colL;
#pragma unroll
    for (int v = 0; v < 8; ++v) {
      int p = mt * 16 + rbase + v, ti = p >> 2, di = p & 3;
      float cc = c_t[(size_t)(b * Tt + t0 + ti) * HID + col] +
                 c_d[(size_t)(b * Dd + d0 + di) * HID + col];
      float hv = fmaxf(acc[v] + cc, 0.f);
      lp[v] += hv * s_w2[col];
    }
  }
#pragma unroll
  for (int off = 1; off <= 8; off <<= 1)
#pragma unroll
    for (int v = 0; v < 8; ++v) lp[v] += __shfl_xor(lp[v], off, 32);
  if (colL == 0)
#pragma unroll
    for (int v = 0; v < 8; ++v) atomicAdd(&s_logit[mt * 16 + rbase + v], lp[v]);
  __syncthreads();
  if (tid < 64) {
    int p = tid, ti = p >> 2, di = p & 3;
    assoc[((size_t)(b * Tt + t0 + ti)) * Dd + d0 + di] = s_logit[p] + p2b[0];
  }
}

// ---------------- host-side launch ----------------
extern "C" void kernel_launch(void* const* d_in, const int* in_sizes, int n_in,
                              void* d_out, int out_size, void* d_ws, size_t ws_size,
                              hipStream_t stream) {
  (void)in_sizes; (void)n_in; (void)out_size; (void)ws_size;
  auto f = [&](int i) { return (const float*)d_in[i]; };
  // pytree flatten order: blocks[0] (22 leaves), blocks[1] (22), then sorted top-level keys
  const int I_CTX1 = 44, I_CTX2 = 46, I_DET1 = 48, I_DET2 = 50, I_FUSE1 = 52, I_FUSE2 = 54,
            I_GP = 56, I_NS1 = 58, I_NS2 = 60, I_P1 = 62, I_P2 = 64, I_QP = 66,
            I_REC1 = 68, I_REC2 = 70, I_SUR1 = 72, I_SUR2 = 74, I_TNB = 76, I_TNG = 77,
            I_GEOM = 78, I_QUALH = 79, I_TIME = 80, I_MASK = 81, I_CTXIN = 82,
            I_DGEOM = 83, I_DQUAL = 84;

  const size_t SEQ = (size_t)NS * Ll * HID;   // 8,388,608 halves per buffer
  _Float16* wsH = (_Float16*)d_ws;
  _Float16* Hb = wsH;                 // conv output h (f16)
  _Float16* Qb = wsH + SEQ;           // q-linear (f16)
  _Float16* Yb = wsH + 2 * SEQ;       // block output / temp hlin target chain
  _Float16* Tb = wsH + 3 * SEQ;       // pre-conv temp
  _Float16* Wp = wsH + 4 * SEQ;       // packed scan weights (both blocks) + W1d + W1f
  _Float16* P4 = Wp + PACK_REG1;
  _Float16* P6 = P4 + 16384;
  float* fbase = (float*)(P6 + 16384);
  float* mS = fbase;
  float* oS = mS + (size_t)NS * HID;
  float* rS = oS + (size_t)NS * HID;
  float* pooled = rS + (size_t)NS * HID;
  float* c_t = pooled + (size_t)NS * HID;
  float* c_d = c_t + (size_t)NS * HID;
  float* qeB = c_d + (size_t)Bb * Dd * HID;

  float* out = (float*)d_out;
  float* oAssoc = out;                          // [B,T,D]
  float* oRec = out + (size_t)Bb * Tt * Dd;     // [B,T,4]
  float* oSur = oRec + (size_t)Bb * Tt * 4;     // [B,T]
  float* oNs  = oSur + (size_t)Bb * Tt;         // [B,T,8]

  // pack weights (f32 -> f16 WMMA fragment order)
  PackSrc ps;
  for (int blk = 0; blk < 2; ++blk) {
    int o = blk * 22;
    ps.w[blk][0] = f(o + 8);   // mw.W
    ps.w[blk][1] = f(o + 6);   // mc.W
    ps.w[blk][2] = f(o + 14);  // ow.W
    ps.w[blk][3] = f(o + 10);  // oc.W
    ps.w[blk][4] = f(o + 20);  // rw.W
    ps.w[blk][5] = f(o + 18);  // rc.W
    ps.w[blk][6] = f(o + 12);  // out.W
  }
  ps.pair1 = f(I_P1);
  k_pack<<<(PACK_TOTAL + 255) / 256, 256, 0, stream>>>(ps, Wp);

  for (int blk = 0; blk < 2; ++blk) {
    int o = blk * 22;
    int inDim = (blk == 0) ? (GEOM + QUAL + TIMEF) : HID;
    k_inlin<<<(NS * Ll + 255) / 256, 256, 0, stream>>>(
        blk, inDim, f(I_GEOM), f(I_QUALH), f(I_TIME), Yb,
        f(o + 5), f(o + 4), f(o + 2), f(o + 3), Tb);
    k_conv<<<(int)((SEQ + 255) / 256), 256, 0, stream>>>(Tb, Hb, f(o + 0), f(o + 1));
    k_qlin<<<(NS * Ll + 255) / 256, 256, 0, stream>>>(f(I_QUALH), f(o + 16), f(o + 17), Qb);
    ScanBias sb;
    sb.b[0] = f(o + 9);  sb.b[1] = f(o + 7);  sb.b[2] = f(o + 15);
    sb.b[3] = f(o + 11); sb.b[4] = f(o + 21); sb.b[5] = f(o + 19); sb.b[6] = f(o + 13);
    k_scan<<<NS / 16, 256, 0, stream>>>(Hb, Qb, Yb, Wp + (size_t)blk * BLK_PACK_HALVES,
                                        sb, f(I_MASK), mS, oS, rS);
  }

  k_pool<<<NS, 64, 0, stream>>>(Yb, f(I_MASK), f(I_TNG), f(I_TNB), pooled);

  TrackArgs ta;
  ta.pooled = pooled; ta.mS = mS; ta.oS = oS; ta.rS = rS; ta.ctx = f(I_CTXIN);
  ta.ctx1W = f(I_CTX1); ta.ctx1b = f(I_CTX1 + 1); ta.ctx2W = f(I_CTX2); ta.ctx2b = f(I_CTX2 + 1);
  ta.f1W = f(I_FUSE1); ta.f1b = f(I_FUSE1 + 1); ta.f2W = f(I_FUSE2); ta.f2b = f(I_FUSE2 + 1);
  ta.ns1W = f(I_NS1); ta.ns1b = f(I_NS1 + 1); ta.ns2W = f(I_NS2); ta.ns2b = f(I_NS2 + 1);
  ta.rec1W = f(I_REC1); ta.rec1b = f(I_REC1 + 1); ta.rec2W = f(I_REC2); ta.rec2b = f(I_REC2 + 1);
  ta.sur1W = f(I_SUR1); ta.sur1b = f(I_SUR1 + 1); ta.sur2W = f(I_SUR2); ta.sur2b = f(I_SUR2 + 1);
  ta.gpW = f(I_GP); ta.gpb = f(I_GP + 1); ta.p1W = f(I_P1); ta.p1b = f(I_P1 + 1);
  ta.outRec = oRec; ta.outSur = oSur; ta.outNs = oNs; ta.c_t = c_t;
  k_track<<<NS, 128, 0, stream>>>(ta);

  k_det<<<Bb * Dd, 128, 0, stream>>>(f(I_DGEOM), f(I_DQUAL),
      f(I_DET1), f(I_DET1 + 1), f(I_DET2), f(I_DET2 + 1),
      f(I_QP), f(I_QP + 1), f(I_GP), f(I_GP + 1), f(I_P1), c_d, qeB);

  k_pair<<<dim3(Tt / 16, Dd / 4, Bb), 256, 0, stream>>>(
      P4, P6, oS, qeB, c_t, c_d, oNs, f(I_DGEOM),
      f(I_GP), f(I_GP + 1), f(I_P2), f(I_P2 + 1), oAssoc);
}
